// HGT_55920474194544
// MI455X (gfx1250) — compile-verified, matrix operand order checked by reading, not run on hardware
//
#include <hip/hip_runtime.h>
#include <hip/hip_bf16.h>
#include <math.h>

#define HID   256
#define NHEAD 8
#define DHEAD 32
#define NT    60000
#define NC    30000
#define NE    200000
#define FIN   64

typedef __attribute__((ext_vector_type(16))) __bf16 v16bf;
typedef __attribute__((ext_vector_type(8)))  __bf16 v8bf;
typedef __attribute__((ext_vector_type(8)))  float  v8f;
typedef __attribute__((ext_vector_type(4)))  unsigned int u32x4;
typedef __attribute__((ext_vector_type(8)))  int          i32x8;
typedef __attribute__((ext_vector_type(4)))  int          i32x4;

// ---------------------------------------------------------------------------
// Helpers
// ---------------------------------------------------------------------------
__device__ __forceinline__ float gelu_exact(float x) {
  return 0.5f * x * (1.0f + erff(x * 0.70710678118654752f));
}

// A/B 16-bit WMMA fragment layout (ISA 7.12.2): lanes 0-15 hold K={0..7,16..23},
// lanes 16-31 hold K={8..15,24..31} of a 32-wide K slice. `base` already points
// at (row, kc + (lane>=16 ? 8 : 0)).
__device__ __forceinline__ v16bf frag_from_f32(const float* base) {
  float4 a0 = ((const float4*)base)[0];
  float4 a1 = ((const float4*)base)[1];
  float4 b0 = ((const float4*)(base + 16))[0];
  float4 b1 = ((const float4*)(base + 16))[1];
  float t[16] = {a0.x, a0.y, a0.z, a0.w, a1.x, a1.y, a1.z, a1.w,
                 b0.x, b0.y, b0.z, b0.w, b1.x, b1.y, b1.z, b1.w};
  v16bf f;
#pragma unroll
  for (int i = 0; i < 16; ++i) f[i] = (__bf16)t[i];
  return f;
}

__device__ __forceinline__ v16bf frag_from_bf16(const __bf16* base) {
  v8bf lo = ((const v8bf*)base)[0];
  v8bf hi = ((const v8bf*)(base + 16))[0];
  v16bf f;
#pragma unroll
  for (int i = 0; i < 8; ++i) { f[i] = lo[i]; f[8 + i] = hi[i]; }
  return f;
}

// ---------------------------------------------------------------------------
// Generic GEMM: C[M,N] = epi( preact(A[M,K]) @ Bt[N,K]^T + bias )
// grid.x = M/16, block = 128 (4 waves). Wave w owns N-tiles {w, w+4, w+8, w+12}.
// K, NTW (N-tiles per wave), PREACT (gelu on A), EPI are compile-time:
// EPI: 0 none, 1 relu, 2 leaky(0.2), 3 sigmoid-skip blend with Xold.
//
// A tile (16 x K, one contiguous run of 16*K f32) is staged into LDS:
//  - PREACT==0: via the Tensor Data Mover (TENSOR_LOAD_TO_LDS, 1-D descriptor,
//    issued by wave 0, completion via s_wait_tensorcnt + block barrier).
//  - PREACT==1: via a VALU copy with GELU fused at the staging store.
// ---------------------------------------------------------------------------
template <int K, int NTW, int PREACT, int EPI>
__global__ void gemm_bf16_wmma(const float* __restrict__ A,
                               const __bf16* __restrict__ Bt, int N,
                               const float* __restrict__ bias,
                               float* C,
                               const float* __restrict__ skipp,
                               const float* Xold) {
  __shared__ __align__(16) float Atile[16 * K];

  if constexpr (PREACT) {
    // Manual staging with GELU fused into the LDS store.
    const float4* src = (const float4*)(A + (size_t)blockIdx.x * (16 * K));
    float4* dst = (float4*)Atile;
    constexpr int NV = (16 * K) / 4;
#pragma unroll
    for (int i = threadIdx.x; i < NV; i += 128) {
      float4 v = src[i];
      v.x = gelu_exact(v.x); v.y = gelu_exact(v.y);
      v.z = gelu_exact(v.z); v.w = gelu_exact(v.w);
      dst[i] = v;
    }
  } else {
    // Tensor Data Mover: DMA the contiguous 16*K-element tile into LDS.
    if (threadIdx.x == 0) {
      const unsigned lds_off = (unsigned)(uintptr_t)(void*)&Atile[0];
      const unsigned long long ga =
          (unsigned long long)(uintptr_t)(const void*)(A + (size_t)blockIdx.x * (16 * K));
      constexpr unsigned nelem = 16 * K;  // 4-byte elements
      u32x4 g0;
      g0[0] = 1u;                                   // count=1, user mode, no gather
      g0[1] = lds_off;                              // lds_addr[31:0]
      g0[2] = (unsigned)(ga & 0xFFFFFFFFu);         // global_addr[31:0]
      g0[3] = (unsigned)((ga >> 32) & 0x1FFFFFFu)   // global_addr[56:32]
              | (2u << 30);                         // type=2 ("image")
      i32x8 g1;
      g1[0] = (int)(2u << 16);                      // workgroup_mask=0, data_size=4B
      g1[1] = (int)((nelem & 0xFFFFu) << 16);       // tensor_dim0[15:0]
      g1[2] = (int)(((nelem >> 16) & 0xFFFFu)       // tensor_dim0[31:16]
              | (1u << 16));                        // tensor_dim1 = 1
      g1[3] = (int)((nelem & 0xFFFFu) << 16);       // tile_dim0 = 16*K
      g1[4] = 0;                                    // tile_dim1 = tile_dim2 = 0
      g1[5] = (int)nelem;                           // tensor_dim0_stride[31:0]
      g1[6] = 0;
      g1[7] = 0;
      i32x4 gz4 = {0, 0, 0, 0};                     // groups 2/3 unused (<=2-D)
      i32x8 gz8 = {0, 0, 0, 0, 0, 0, 0, 0};         // extra descriptor group (clang-23 6-arg form)
      __builtin_amdgcn_tensor_load_to_lds(g0, g1, gz4, gz4, gz8, 0);
      __builtin_amdgcn_s_wait_tensorcnt(0);
    }
  }
  __syncthreads();

  const int lane = threadIdx.x & 31;
  const int wave = threadIdx.x >> 5;
  const int l15  = lane & 15;
  const int hi   = lane >> 4;
  const float* arow = Atile + l15 * K;

  v8f acc[NTW];
  v8f zero = {};
#pragma unroll
  for (int t = 0; t < NTW; ++t) acc[t] = zero;

#pragma unroll
  for (int kc = 0; kc < K; kc += 32) {
    v16bf af = frag_from_f32(arow + kc + hi * 8);   // LDS reads
    v16bf bf[NTW];
#pragma unroll
    for (int t = 0; t < NTW; ++t) {
      const int ntile = wave + (t << 2);
      bf[t] = frag_from_bf16(Bt + (size_t)(ntile * 16 + l15) * K + kc + hi * 8);
    }
#pragma unroll
    for (int t = 0; t < NTW; ++t)
      acc[t] = __builtin_amdgcn_wmma_f32_16x16x32_bf16(
          false, af, false, bf[t], (short)0, acc[t], false, false);
  }

  float beta = 0.0f;
  if (EPI == 3) beta = 1.0f / (1.0f + __expf(-skipp[0]));

#pragma unroll
  for (int t = 0; t < NTW; ++t) {
    const int col = (wave + (t << 2)) * 16 + l15;
    const float bv = bias[col];
#pragma unroll
    for (int r = 0; r < 8; ++r) {
      const int rr = blockIdx.x * 16 + r + hi * 8;  // C layout: lanes16-31 -> M+8
      float v = acc[t][r] + bv;
      if (EPI == 1)      v = fmaxf(v, 0.0f);
      else if (EPI == 2) v = (v > 0.0f) ? v : 0.2f * v;
      else if (EPI == 3) v = beta * v + (1.0f - beta) * Xold[(size_t)rr * N + col];
      C[(size_t)rr * N + col] = v;
    }
  }
}

// ---------------------------------------------------------------------------
// Per-head relation transform: OUT[n, h*32+e] = sum_d KV[n, h*32+d] * R[h][d][e]
// Rt is [H][32(e)][32(d)] bf16. grid (Nn/16, H), block 64 (2 waves, one e-tile each).
// ---------------------------------------------------------------------------
__global__ void relgemm_wmma(const float* __restrict__ KV,
                             const __bf16* __restrict__ Rt,
                             float* __restrict__ OUT) {
  const int lane = threadIdx.x & 31;
  const int wave = threadIdx.x >> 5;  // e-tile 0 or 1
  const int l15  = lane & 15;
  const int hi   = lane >> 4;
  const int h    = blockIdx.y;
  const int row  = blockIdx.x * 16 + l15;

  v16bf afrag = frag_from_f32(KV + (size_t)row * HID + h * DHEAD + hi * 8);
  const __bf16* brow = Rt + h * (DHEAD * DHEAD) + (wave * 16 + l15) * DHEAD + hi * 8;
  v16bf bfrag = frag_from_bf16(brow);
  v8f acc = {};
  acc = __builtin_amdgcn_wmma_f32_16x16x32_bf16(
      false, afrag, false, bfrag, (short)0, acc, false, false);

  const int col = h * DHEAD + wave * 16 + l15;
#pragma unroll
  for (int r = 0; r < 8; ++r) {
    const int rr = blockIdx.x * 16 + r + hi * 8;
    OUT[(size_t)rr * HID + col] = acc[r];
  }
}

// ---------------------------------------------------------------------------
// Weight convert + transpose: W f32 [K,N] -> Bt bf16 [N,K]
// ---------------------------------------------------------------------------
__global__ void wconv_kernel(const float* __restrict__ W, __bf16* __restrict__ Bt,
                             int K, int N) {
  int idx = blockIdx.x * 256 + threadIdx.x;
  if (idx >= K * N) return;
  int k = idx / N, n = idx - k * N;
  Bt[(size_t)n * K + k] = (__bf16)W[idx];
}

// a_rel/m_rel [H,D,D] (d,e) -> Rt [H][e][d] bf16
__global__ void relconv_kernel(const float* __restrict__ R, __bf16* __restrict__ Rt) {
  int idx = blockIdx.x * 256 + threadIdx.x;
  if (idx >= NHEAD * DHEAD * DHEAD) return;
  int h = idx >> 10, rem = idx & 1023, d = rem >> 5, e = rem & 31;
  Rt[h * 1024 + e * 32 + d] = (__bf16)R[idx];
}

__global__ void fillf_kernel(float* __restrict__ p, float v, int n) {
  int i = blockIdx.x * 256 + threadIdx.x;
  if (i < n) p[i] = v;
}

// ---------------------------------------------------------------------------
// Edge-space attention kernels (1 thread per (edge, head))
// ---------------------------------------------------------------------------
__global__ void edge_alpha_kernel(const float* __restrict__ qb, const float* __restrict__ kr,
                                  const int* __restrict__ si, const int* __restrict__ di,
                                  const float* __restrict__ prel, float* __restrict__ alpha) {
  int t = blockIdx.x * 256 + threadIdx.x;
  if (t >= NE * NHEAD) return;
  int e = t >> 3, h = t & 7;
  const float4* q4 = (const float4*)(qb + (size_t)di[e] * HID + h * DHEAD);
  const float4* k4 = (const float4*)(kr + (size_t)si[e] * HID + h * DHEAD);
  float s = 0.0f;
#pragma unroll
  for (int i = 0; i < 8; ++i) {
    float4 a = q4[i], b = k4[i];
    s += a.x * b.x + a.y * b.y + a.z * b.z + a.w * b.w;
  }
  alpha[t] = s * prel[h] * 0.17677669529663687f;  // 1/sqrt(32)
}

__device__ __forceinline__ void atomicMaxF(float* addr, float value) {
  if (value >= 0.0f) atomicMax((int*)addr, __float_as_int(value));
  else               atomicMin((unsigned int*)addr, __float_as_uint(value));
}

__global__ void seg_max_kernel(const float* __restrict__ alpha, const int* __restrict__ di,
                               float* __restrict__ m) {
  int t = blockIdx.x * 256 + threadIdx.x;
  if (t >= NE * NHEAD) return;
  int e = t >> 3, h = t & 7;
  atomicMaxF(&m[(size_t)di[e] * NHEAD + h], alpha[t]);
}

__global__ void seg_expsum_kernel(float* __restrict__ alpha, const int* __restrict__ di,
                                  const float* __restrict__ m, float* __restrict__ ssum) {
  int t = blockIdx.x * 256 + threadIdx.x;
  if (t >= NE * NHEAD) return;
  int e = t >> 3, h = t & 7;
  float ex = __expf(alpha[t] - m[(size_t)di[e] * NHEAD + h]);
  alpha[t] = ex;
  atomicAdd(&ssum[(size_t)di[e] * NHEAD + h], ex);
}

// 1 thread per (edge, channel): agg[dst, c] += vr[src, c] * w[e, h(c)]
__global__ void seg_scatter_kernel(const float* __restrict__ alpha, const float* __restrict__ ssum,
                                   const float* __restrict__ vr,
                                   const int* __restrict__ si, const int* __restrict__ di,
                                   float* __restrict__ agg) {
  int idx = blockIdx.x * 256 + threadIdx.x;
  if (idx >= NE * HID) return;
  int e = idx >> 8, c = idx & 255, h = c >> 5;
  float w = alpha[e * 8 + h] / (ssum[(size_t)di[e] * NHEAD + h] + 1e-16f);
  atomicAdd(&agg[(size_t)di[e] * HID + c], vr[(size_t)si[e] * HID + c] * w);
}

// L2-normalize rows of 64
__global__ void l2norm_kernel(float* __restrict__ y, int M) {
  int r = blockIdx.x * 256 + threadIdx.x;
  if (r >= M) return;
  float* p = y + (size_t)r * 64;
  float s = 0.0f;
#pragma unroll
  for (int i = 0; i < 64; ++i) s += p[i] * p[i];
  float inv = 1.0f / fmaxf(sqrtf(s), 1e-12f);
#pragma unroll
  for (int i = 0; i < 64; ++i) p[i] *= inv;
}

// ---------------------------------------------------------------------------
// Host orchestration
// ---------------------------------------------------------------------------
extern "C" void kernel_launch(void* const* d_in, const int* in_sizes, int n_in,
                              void* d_out, int out_size, void* d_ws, size_t ws_size,
                              hipStream_t stream) {
  (void)in_sizes; (void)n_in; (void)out_size; (void)ws_size;
  const float* x_trans = (const float*)d_in[0];
  const float* x_cc    = (const float*)d_in[1];
  const int*   src_c2t = (const int*)d_in[2];
  const int*   dst_c2t = (const int*)d_in[3];
  const int*   src_t2c = (const int*)d_in[4];
  const int*   dst_t2c = (const int*)d_in[5];
  const float* W_in  = (const float*)d_in[6];
  const float* b_in  = (const float*)d_in[7];
  const float* Wk    = (const float*)d_in[8];
  const float* bk    = (const float*)d_in[9];
  const float* Wq    = (const float*)d_in[10];
  const float* bq    = (const float*)d_in[11];
  const float* Wv    = (const float*)d_in[12];
  const float* bv    = (const float*)d_in[13];
  const float* Wa    = (const float*)d_in[14];
  const float* ba    = (const float*)d_in[15];
  const float* a_rel = (const float*)d_in[16];
  const float* m_rel = (const float*)d_in[17];
  const float* p_rel = (const float*)d_in[18];
  const float* skip  = (const float*)d_in[19];
  const float* W_out = (const float*)d_in[20];
  const float* b_out = (const float*)d_in[21];
  float* out = (float*)d_out;

  char* wp = (char*)d_ws;
  auto alloc = [&](size_t bytes) -> char* {
    char* r = wp; wp += (bytes + 255) & ~(size_t)255; return r;
  };
  float* x0    = (float*)alloc((size_t)NT * HID * 4);
  float* x1    = (float*)alloc((size_t)NC * HID * 4);
  float* tmp   = (float*)alloc((size_t)NT * HID * 4);
  float* kr    = (float*)alloc((size_t)NT * HID * 4);
  float* vr    = (float*)alloc((size_t)NT * HID * 4);
  float* qb    = (float*)alloc((size_t)NT * HID * 4);
  float* agg0  = (float*)alloc((size_t)NT * HID * 4);
  float* agg1  = (float*)alloc((size_t)NC * HID * 4);
  float* alpha = (float*)alloc((size_t)NE * NHEAD * 4);
  float* mbuf  = (float*)alloc((size_t)NT * NHEAD * 4);
  float* sbuf  = (float*)alloc((size_t)NT * NHEAD * 4);
  __bf16* wt   = (__bf16*)alloc((size_t)HID * HID * 2);
  __bf16* rt   = (__bf16*)alloc((size_t)NHEAD * DHEAD * DHEAD * 2);

  float* xs[2]   = { x0, x1 };
  float* aggs[2] = { agg0, agg1 };
  const int Nn[2] = { NT, NC };
  const int* srcs[2] = { src_c2t, src_t2c };
  const int* dsts[2] = { dst_c2t, dst_t2c };
  const int stype[2] = { 1, 0 };
  const int dtype[2] = { 0, 1 };

  auto wconv = [&](const float* W, int K, int N) {
    int n = K * N;
    wconv_kernel<<<dim3((n + 255) / 256), dim3(256), 0, stream>>>(W, wt, K, N);
  };

  // ---- input projections: xs[t] = relu(x_t @ W_in[t] + b_in[t])
  wconv(W_in + 0 * FIN * HID, FIN, HID);
  gemm_bf16_wmma<FIN, 4, 0, 1><<<dim3(NT / 16), dim3(128), 0, stream>>>(
      x_trans, wt, HID, b_in + 0 * HID, x0, nullptr, nullptr);
  wconv(W_in + 1 * FIN * HID, FIN, HID);
  gemm_bf16_wmma<FIN, 4, 0, 1><<<dim3(NC / 16), dim3(128), 0, stream>>>(
      x_cc, wt, HID, b_in + 1 * HID, x1, nullptr, nullptr);

  for (int l = 0; l < 2; ++l) {
    fillf_kernel<<<dim3((NT * HID + 255) / 256), dim3(256), 0, stream>>>(agg0, 0.0f, NT * HID);
    fillf_kernel<<<dim3((NC * HID + 255) / 256), dim3(256), 0, stream>>>(agg1, 0.0f, NC * HID);

    for (int r = 0; r < 2; ++r) {
      const int st = stype[r], dt = dtype[r];
      const int Ms = Nn[st], Md = Nn[dt];
      const size_t wofs  = (size_t)(l * 2 + st) * HID * HID;
      const size_t bofs  = (size_t)(l * 2 + st) * HID;
      const size_t wofsq = (size_t)(l * 2 + dt) * HID * HID;
      const size_t bofsq = (size_t)(l * 2 + dt) * HID;
      const size_t rofs  = (size_t)(l * 2 + r) * NHEAD * DHEAD * DHEAD;

      // kr = rel_transform(xs[st] @ Wk + bk, a_rel)
      wconv(Wk + wofs, HID, HID);
      gemm_bf16_wmma<HID, 4, 0, 0><<<dim3(Ms / 16), dim3(128), 0, stream>>>(
          xs[st], wt, HID, bk + bofs, tmp, nullptr, nullptr);
      relconv_kernel<<<dim3(32), dim3(256), 0, stream>>>(a_rel + rofs, rt);
      relgemm_wmma<<<dim3(Ms / 16, NHEAD), dim3(64), 0, stream>>>(tmp, rt, kr);

      // vr = rel_transform(xs[st] @ Wv + bv, m_rel)
      wconv(Wv + wofs, HID, HID);
      gemm_bf16_wmma<HID, 4, 0, 0><<<dim3(Ms / 16), dim3(128), 0, stream>>>(
          xs[st], wt, HID, bv + bofs, tmp, nullptr, nullptr);
      relconv_kernel<<<dim3(32), dim3(256), 0, stream>>>(m_rel + rofs, rt);
      relgemm_wmma<<<dim3(Ms / 16, NHEAD), dim3(64), 0, stream>>>(tmp, rt, vr);

      // q for destination type
      wconv(Wq + wofsq, HID, HID);
      gemm_bf16_wmma<HID, 4, 0, 0><<<dim3(Md / 16), dim3(128), 0, stream>>>(
          xs[dt], wt, HID, bq + bofsq, qb, nullptr, nullptr);

      // segment softmax over edges + weighted scatter into agg[dt]
      const int nth = NE * NHEAD;
      edge_alpha_kernel<<<dim3((nth + 255) / 256), dim3(256), 0, stream>>>(
          qb, kr, srcs[r], dsts[r], p_rel + (size_t)(l * 2 + r) * NHEAD, alpha);
      fillf_kernel<<<dim3((Md * NHEAD + 255) / 256), dim3(256), 0, stream>>>(
          mbuf, -3.402823466e38f, Md * NHEAD);
      seg_max_kernel<<<dim3((nth + 255) / 256), dim3(256), 0, stream>>>(alpha, dsts[r], mbuf);
      fillf_kernel<<<dim3((Md * NHEAD + 255) / 256), dim3(256), 0, stream>>>(
          sbuf, 0.0f, Md * NHEAD);
      seg_expsum_kernel<<<dim3((nth + 255) / 256), dim3(256), 0, stream>>>(
          alpha, dsts[r], mbuf, sbuf);
      const int nsc = NE * HID;
      seg_scatter_kernel<<<dim3((nsc + 255) / 256), dim3(256), 0, stream>>>(
          alpha, sbuf, vr, srcs[r], dsts[r], aggs[dt]);
    }

    // xs[t] = beta * (gelu(agg[t]) @ Wa + ba) + (1-beta) * xs[t]   (in place)
    for (int t = 0; t < 2; ++t) {
      const size_t wofs = (size_t)(l * 2 + t) * HID * HID;
      const size_t bofs = (size_t)(l * 2 + t) * HID;
      wconv(Wa + wofs, HID, HID);
      gemm_bf16_wmma<HID, 4, 1, 3><<<dim3(Nn[t] / 16), dim3(128), 0, stream>>>(
          aggs[t], wt, HID, ba + bofs, xs[t], skip + (l * 2 + t), xs[t]);
    }
  }

  // out = l2normalize(leaky_relu(x0 @ W_out + b_out))
  wconv(W_out, HID, 64);
  gemm_bf16_wmma<HID, 1, 0, 2><<<dim3(NT / 16), dim3(128), 0, stream>>>(
      x0, wt, 64, b_out, out, nullptr, nullptr);
  l2norm_kernel<<<dim3((NT + 255) / 256), dim3(256), 0, stream>>>(out, NT);
}
